// ChebNet_NC_43542378447164
// MI455X (gfx1250) — matrix operand order repaired and checked by reading, not compiled, not run on hardware
//
#include <hip/hip_runtime.h>
#include <hip/hip_fp16.h>

typedef _Float16 f16;
typedef __attribute__((ext_vector_type(16))) _Float16 v16h;
typedef __attribute__((ext_vector_type(8)))  float    v8f;

static __device__ __forceinline__ int gtid() {
  return blockIdx.x * blockDim.x + threadIdx.x;
}

// ---------------- elementwise / graph kernels ----------------

__global__ void k_zero(float* __restrict__ p, int n) {
  int t = gtid();
  if (t < n) p[t] = 0.f;
}

__global__ void k_cvt_f16(const float* __restrict__ s, f16* __restrict__ d, int n) {
  int t = gtid();
  if (t < n) d[t] = (f16)s[t];
}

__global__ void k_deg(const int* __restrict__ row, float* __restrict__ deg, int e) {
  int t = gtid();
  if (t < e) atomicAdd(&deg[row[t]], 1.0f);
}

__global__ void k_dinv(const float* __restrict__ deg, float* __restrict__ dinv, int n) {
  int t = gtid();
  if (t < n) {
    float d = deg[t];
    dinv[t] = d > 0.f ? rsqrtf(fmaxf(d, 1.f)) : 0.f;
  }
}

__global__ void k_wedge(const int* __restrict__ row, const int* __restrict__ col,
                        const float* __restrict__ dinv, float* __restrict__ w, int e) {
  int t = gtid();
  if (t < e) w[t] = -dinv[row[t]] * dinv[col[t]];
}

// Edge-parallel SpMM: one thread per (edge, feature-quad). Gather hits L2
// (whole feature matrix resident); scatter via global_atomic_add_f32.
__global__ void k_prop(const float* __restrict__ src, const int* __restrict__ row,
                       const int* __restrict__ col, const float* __restrict__ w,
                       float* __restrict__ dst, int etot) {
  int t = gtid();
  if (t >= etot) return;
  int ed = t >> 4, q = (t & 15) << 2;
  float we = w[ed];
  const float4 v = *reinterpret_cast<const float4*>(src + (size_t)col[ed] * 64 + q);
  float* d = dst + (size_t)row[ed] * 64 + q;
  atomicAdd(d + 0, we * v.x);
  atomicAdd(d + 1, we * v.y);
  atomicAdd(d + 2, we * v.z);
  atomicAdd(d + 3, we * v.w);
}

// Chebyshev recurrence tail: d = 2*d - s (d holds prop(Tx1))
__global__ void k_t2(float* __restrict__ d, const float* __restrict__ s, int n) {
  int t = gtid();
  if (t < n) d[t] = 2.f * d[t] - s[t];
}

// ---------------- WMMA GEMM kernels ----------------
// out[M,64] = relu([S0|S1|S2][M,192] @ Bp[192,64] + bias)
// One wave = one 16-col slab (nt = wid&3), grid-strided over 16-row M tiles.
__global__ void k_gemm_relu(const float* __restrict__ S0, const float* __restrict__ S1,
                            const float* __restrict__ S2, const f16* __restrict__ Bp,
                            const float* __restrict__ bias, float* __restrict__ out,
                            int mtiles) {
  const int lane = threadIdx.x & 31;
  const int wpb  = blockDim.x >> 5;
  const int wid  = blockIdx.x * wpb + (threadIdx.x >> 5);
  const int nw   = gridDim.x * wpb;
  const int ln   = lane & 15;
  const int hh   = lane >> 4;
  const int nt   = wid & 3;

  // Load the 6 B fragments once per wave (B: lane=col ln, K = 16*hh + t)
  v16h Bf[6];
#pragma unroll
  for (int c = 0; c < 6; ++c) {
    const f16* bp = Bp + (size_t)(32 * c + 16 * hh) * 64 + nt * 16 + ln;
    v16h tmp;
#pragma unroll
    for (int t = 0; t < 16; ++t) tmp[t] = bp[(size_t)t * 64];
    Bf[c] = tmp;
  }
  const float bv = bias[nt * 16 + ln];
  const float* srcs[3] = {S0, S1, S2};

  for (int mt = wid >> 2; mt < mtiles; mt += (nw >> 2)) {
    v8f acc = {};
#pragma unroll
    for (int c = 0; c < 6; ++c) {
      const float* s = srcs[c >> 1];
      const int kb = (c & 1) * 32;
      v16h Af;
#pragma unroll
      for (int g = 0; g < 2; ++g) {
        // A: lane holds row ln, contiguous K-run of 8 at kb + 16*hh + 8*g
        const float* p = s + (size_t)(mt * 16 + ln) * 64 + kb + 16 * hh + 8 * g;
        float4 u = *reinterpret_cast<const float4*>(p);
        float4 v = *reinterpret_cast<const float4*>(p + 4);
        Af[8 * g + 0] = (f16)u.x; Af[8 * g + 1] = (f16)u.y;
        Af[8 * g + 2] = (f16)u.z; Af[8 * g + 3] = (f16)u.w;
        Af[8 * g + 4] = (f16)v.x; Af[8 * g + 5] = (f16)v.y;
        Af[8 * g + 6] = (f16)v.z; Af[8 * g + 7] = (f16)v.w;
      }
      acc = __builtin_amdgcn_wmma_f32_16x16x32_f16(false, Af, false, Bf[c],
                                                   (short)0, acc, false, false);
    }
    // fused bias + relu store (C/D: VGPR r -> row r+8*hh, col ln)
#pragma unroll
    for (int r = 0; r < 8; ++r) {
      int mrow = mt * 16 + r + 8 * hh;
      float v = acc[r] + bv;
      out[(size_t)mrow * 64 + nt * 16 + ln] = v > 0.f ? v : 0.f;
    }
  }
}

// out[M,16] = log_softmax([S0|S1|S2][M,192] @ Bp[192,16] + bias, axis=1)
// C=16 => one accumulator VGPR holds a full row per 16-lane half: reduce with shfl_xor.
__global__ void k_gemm_lsm(const float* __restrict__ S0, const float* __restrict__ S1,
                           const float* __restrict__ S2, const f16* __restrict__ Bp,
                           const float* __restrict__ bias, float* __restrict__ out,
                           int mtiles) {
  const int lane = threadIdx.x & 31;
  const int wpb  = blockDim.x >> 5;
  const int wid  = blockIdx.x * wpb + (threadIdx.x >> 5);
  const int nw   = gridDim.x * wpb;
  const int ln   = lane & 15;
  const int hh   = lane >> 4;

  v16h Bf[6];
#pragma unroll
  for (int c = 0; c < 6; ++c) {
    const f16* bp = Bp + (size_t)(32 * c + 16 * hh) * 16 + ln;
    v16h tmp;
#pragma unroll
    for (int t = 0; t < 16; ++t) tmp[t] = bp[(size_t)t * 16];
    Bf[c] = tmp;
  }
  const float bv = bias[ln];
  const float* srcs[3] = {S0, S1, S2};

  for (int mt = wid; mt < mtiles; mt += nw) {
    v8f acc = {};
#pragma unroll
    for (int c = 0; c < 6; ++c) {
      const float* s = srcs[c >> 1];
      const int kb = (c & 1) * 32;
      v16h Af;
#pragma unroll
      for (int g = 0; g < 2; ++g) {
        const float* p = s + (size_t)(mt * 16 + ln) * 64 + kb + 16 * hh + 8 * g;
        float4 u = *reinterpret_cast<const float4*>(p);
        float4 v = *reinterpret_cast<const float4*>(p + 4);
        Af[8 * g + 0] = (f16)u.x; Af[8 * g + 1] = (f16)u.y;
        Af[8 * g + 2] = (f16)u.z; Af[8 * g + 3] = (f16)u.w;
        Af[8 * g + 4] = (f16)v.x; Af[8 * g + 5] = (f16)v.y;
        Af[8 * g + 6] = (f16)v.z; Af[8 * g + 7] = (f16)v.w;
      }
      acc = __builtin_amdgcn_wmma_f32_16x16x32_f16(false, Af, false, Bf[c],
                                                   (short)0, acc, false, false);
    }
    // fused bias + log_softmax per row (16 values live across one 16-lane half)
#pragma unroll
    for (int r = 0; r < 8; ++r) {
      float v = acc[r] + bv;
      float mx = v;
#pragma unroll
      for (int d = 1; d < 16; d <<= 1) mx = fmaxf(mx, __shfl_xor(mx, d, 32));
      float ex = __expf(v - mx);
      float sm = ex;
#pragma unroll
      for (int d = 1; d < 16; d <<= 1) sm += __shfl_xor(sm, d, 32);
      out[(size_t)(mt * 16 + r + 8 * hh) * 16 + ln] = v - mx - __logf(sm);
    }
  }
}

// ---------------- launch ----------------

extern "C" void kernel_launch(void* const* d_in, const int* in_sizes, int n_in,
                              void* d_out, int out_size, void* d_ws, size_t ws_size,
                              hipStream_t stream) {
  const float* x  = (const float*)d_in[0];
  const int*   ei = (const int*)d_in[1];
  const float* W1 = (const float*)d_in[2];
  const float* b1 = (const float*)d_in[3];
  const float* W2 = (const float*)d_in[4];
  const float* b2 = (const float*)d_in[5];
  float* out = (float*)d_out;

  const int n = 100000;
  const int e = in_sizes[1] / 2;  // 1600000
  const int* row = ei;
  const int* col = ei + e;

  char* base = (char*)d_ws;
  size_t off = 0;
  auto take = [&](size_t bytes) {
    char* p = base + off;
    off = (off + bytes + 255) & ~(size_t)255;
    return p;
  };
  float* deg  = (float*)take((size_t)n * 4);
  float* dinv = (float*)take((size_t)n * 4);
  float* wed  = (float*)take((size_t)e * 4);
  float* T1   = (float*)take((size_t)n * 64 * 4);  // Tx1 / S1
  float* T2   = (float*)take((size_t)n * 64 * 4);  // Tx2 / S2
  float* H    = (float*)take((size_t)n * 64 * 4);  // layer-1 activations
  f16*   Bp1  = (f16*)take(192 * 64 * 2);
  f16*   Bp2  = (f16*)take(192 * 16 * 2);

  const int TB = 256;
  auto G = [](int nn) { return (nn + 255) / 256; };
  const int nf   = n * 64;
  const int etot = e * 16;
  const int mtiles = n / 16;  // 6250

  // graph normalization
  k_zero <<<G(n), TB, 0, stream>>>(deg, n);
  k_deg  <<<G(e), TB, 0, stream>>>(row, deg, e);
  k_dinv <<<G(n), TB, 0, stream>>>(deg, dinv, n);
  k_wedge<<<G(e), TB, 0, stream>>>(row, col, dinv, wed, e);

  // weight packs (flat (3,64,O) is already the stacked [192,O] B matrix)
  k_cvt_f16<<<G(192 * 64), TB, 0, stream>>>(W1, Bp1, 192 * 64);
  k_cvt_f16<<<G(192 * 16), TB, 0, stream>>>(W2, Bp2, 192 * 16);

  // layer 1 Chebyshev basis
  k_zero<<<G(nf), TB, 0, stream>>>(T1, nf);
  k_prop<<<G(etot), TB, 0, stream>>>(x, row, col, wed, T1, etot);   // Tx1
  k_zero<<<G(nf), TB, 0, stream>>>(T2, nf);
  k_prop<<<G(etot), TB, 0, stream>>>(T1, row, col, wed, T2, etot);  // prop(Tx1)
  k_t2  <<<G(nf), TB, 0, stream>>>(T2, x, nf);                      // Tx2 = 2p - x

  // layer 1 fused GEMM (+bias+relu)
  k_gemm_relu<<<128, 256, 0, stream>>>(x, T1, T2, Bp1, b1, H, mtiles);

  // layer 2 Chebyshev basis
  k_zero<<<G(nf), TB, 0, stream>>>(T1, nf);
  k_prop<<<G(etot), TB, 0, stream>>>(H, row, col, wed, T1, etot);   // S1
  k_zero<<<G(nf), TB, 0, stream>>>(T2, nf);
  k_prop<<<G(etot), TB, 0, stream>>>(T1, row, col, wed, T2, etot);  // prop(S1)
  k_t2  <<<G(nf), TB, 0, stream>>>(T2, H, nf);                      // S2 = 2p - H

  // layer 2 fused GEMM (+bias+log_softmax)
  k_gemm_lsm<<<128, 256, 0, stream>>>(H, T1, T2, Bp2, b2, out, mtiles);
}
